// TransformerBlock_15857019257410
// MI455X (gfx1250) — compile-verified, matrix operand order checked by reading, not run on hardware
//
#include <hip/hip_runtime.h>

typedef __attribute__((ext_vector_type(16))) __bf16 v16bf;
typedef __attribute__((ext_vector_type(8)))  float  v8f;
typedef __attribute__((ext_vector_type(2)))  float  v2f;

constexpr int BN   = 4;
constexpr int SL   = 4096;
constexpr int DIM  = 256;          // D == T == 256
constexpr int NROW = BN * SL;      // 16384 total rows
constexpr int KNN  = 16;
constexpr int XST  = DIM + 3;      // 259: row stride of x (pos3 + feat256)
constexpr int DIMSQ = DIM * DIM;

// ---- gfx1250 async global->LDS path (guarded: falls back to plain copies) ----
#if defined(__HIP_DEVICE_COMPILE__) &&                                         \
    __has_builtin(__builtin_amdgcn_global_load_async_to_lds_b32) &&            \
    __has_builtin(__builtin_amdgcn_global_load_async_to_lds_b128) &&           \
    __has_builtin(__builtin_amdgcn_s_wait_asynccnt)
#define HAVE_ASYNC_LDS 1
#else
#define HAVE_ASYNC_LDS 0
#endif

#if HAVE_ASYNC_LDS
typedef int v4i_vec __attribute__((vector_size(16)));
typedef __attribute__((address_space(1))) int      GI32;
typedef __attribute__((address_space(3))) int      LI32;
typedef __attribute__((address_space(1))) v4i_vec  GI128;
typedef __attribute__((address_space(3))) v4i_vec  LI128;
__device__ inline void async_g2l_b32(const void* g, void* l) {
  __builtin_amdgcn_global_load_async_to_lds_b32((GI32*)g, (LI32*)l, 0, 0);
}
__device__ inline void async_g2l_b128(const void* g, void* l) {
  __builtin_amdgcn_global_load_async_to_lds_b128((GI128*)g, (LI128*)l, 0, 0);
}
__device__ inline void async_wait0() { __builtin_amdgcn_s_wait_asynccnt(0); }
#endif

union Frag16 { uint4 u[2]; v16bf v; };

__device__ inline v8f wmma_bf16(v16bf a, v16bf b, v8f c) {
  return __builtin_amdgcn_wmma_f32_16x16x32_bf16(false, a, false, b, (short)0, c,
                                                 false, false);
}

// A fragment (16x32 bf16) from row-major matrix, row stride DIM
__device__ inline v16bf load_afrag(const __bf16* A, int m0, int kt, int lane) {
  int ll = lane & 15, hi = lane >> 4;
  const __bf16* p = A + (size_t)(m0 + ll) * DIM + kt * 32 + 8 * hi;
  Frag16 f;
  f.u[0] = *(const uint4*)p;
  f.u[1] = *(const uint4*)(p + 16);
  return f.v;
}

// B fragment (32x16 bf16) from pre-swizzled weight buffer
__device__ inline v16bf load_bfrag(const __bf16* Wsw, int kt, int nt, int lane) {
  const __bf16* p = Wsw + ((size_t)(kt * 16 + nt)) * 512 + lane * 16;
  Frag16 f;
  f.u[0] = ((const uint4*)p)[0];
  f.u[1] = ((const uint4*)p)[1];
  return f.v;
}

// ---------------- prep: swizzle 7 weight matrices (f32 row-major -> bf16 B-frag order)
__global__ __launch_bounds__(32) void prep_weights(
    const float* w0, const float* w1, const float* w2, const float* w3,
    const float* w4, const float* w5, const float* w6, __bf16* wsw) {
  int lane = threadIdx.x;
  int f = blockIdx.x;            // kt*16 + nt, 0..127
  int m = blockIdx.y;            // 0..6
  const float* W = m == 0 ? w0 : m == 1 ? w1 : m == 2 ? w2 : m == 3 ? w3
                 : m == 4 ? w4 : m == 5 ? w5 : w6;
  int kt = f >> 4, nt = f & 15;
  int ll = lane & 15, hi = lane >> 4;
  __bf16* dst = wsw + (size_t)m * DIMSQ + (size_t)f * 512 + lane * 16;
  int n = nt * 16 + ll;
#pragma unroll
  for (int e = 0; e < 16; ++e) {
    int k = kt * 32 + (e & 7) + 8 * hi + 16 * (e >> 3);
    dst[e] = (__bf16)W[(size_t)k * DIM + n];
  }
}

// ---------------- prep: pos copy, squared norms, bf16 feat
__global__ __launch_bounds__(32) void prep_feat(const float* x, float* sq,
                                                __bf16* featB, float* posOut) {
  int row = blockIdx.x;
  int lane = threadIdx.x;
  const float* xr = x + (size_t)row * XST;
  float s = 0.f;
#pragma unroll
  for (int t = 0; t < 8; ++t) {
    float v = xr[3 + lane + t * 32];
    s += v * v;
    featB[(size_t)row * DIM + lane + t * 32] = (__bf16)v;
  }
#pragma unroll
  for (int off = 16; off; off >>= 1) s += __shfl_xor(s, off, 32);
  if (lane == 0) sq[row] = s;
  if (lane < 3) posOut[(size_t)row * 3 + lane] = xr[lane];
}

// ---------------- KNN: f32 WMMA Gram tiles + register top-16 selection
__global__ __launch_bounds__(128) void knn_kernel(const float* x, const float* sq,
                                                  int* idx) {
  __shared__ float qs[32 * 256];
  __shared__ float dt[4][512];
  __shared__ float listD[4][32][16];
  __shared__ int   listJ[4][32][16];

  int tid = threadIdx.x;
  int lane = tid & 31;
  int w = tid >> 5;
  int blk = blockIdx.x;                 // 0..511
  int b = blk / (SL / 32);
  int i0 = (blk % (SL / 32)) * 32;
  const float* xb = x + (size_t)b * SL * XST;

  {  // stage 32 query rows into LDS
    int row = tid >> 2;
    int seg = (tid & 3) * 64;
    const float* src = xb + (size_t)(i0 + row) * XST + 3 + seg;
    float* dst = qs + row * 256 + seg;
#if HAVE_ASYNC_LDS
#pragma unroll
    for (int t = 0; t < 64; ++t) async_g2l_b32(src + t, dst + t);
    async_wait0();
#else
    for (int t = 0; t < 64; ++t) dst[t] = src[t];
#endif
  }
  __syncthreads();

  float bd[16];
  int bj[16];
#pragma unroll
  for (int t = 0; t < 16; ++t) { bd[t] = 3.4e38f; bj[t] = 0x7fffffff; }

  int ll = lane & 15, hi = lane >> 4;
  for (int jt = w * 64; jt < w * 64 + 64; ++jt) {
    int j0 = jt * 16;
    v8f c0 = {}, c1 = {};
    const float* brow = xb + (size_t)(j0 + ll) * XST + 3;
    for (int k0 = 0; k0 < 256; k0 += 4) {
      int kk = k0 + 2 * hi;
      v2f a0, a1, bb;
      a0.x = qs[ll * 256 + kk];        a0.y = qs[ll * 256 + kk + 1];
      a1.x = qs[(ll + 16) * 256 + kk]; a1.y = qs[(ll + 16) * 256 + kk + 1];
      bb.x = brow[kk];                 bb.y = brow[kk + 1];
      c0 = __builtin_amdgcn_wmma_f32_16x16x4_f32(false, a0, false, bb, (short)0, c0,
                                                 false, false);
      c1 = __builtin_amdgcn_wmma_f32_16x16x4_f32(false, a1, false, bb, (short)0, c1,
                                                 false, false);
    }
    // dist ordering key: sq_j - 2*dot  (sq_i constant per row, dropped)
    float sqj = sq[b * SL + j0 + ll];
#pragma unroll
    for (int r = 0; r < 8; ++r) {
      dt[w][(r + 8 * hi) * 16 + ll]      = sqj - 2.0f * c0[r];
      dt[w][(16 + r + 8 * hi) * 16 + ll] = sqj - 2.0f * c1[r];
    }
    __syncthreads();   // uniform across all 4 waves (64 iters each)
    float dmax = bd[15];
#pragma unroll
    for (int t = 0; t < 16; ++t) {
      float dc = dt[w][lane * 16 + t];
      if (dc < dmax) {
        float dp = dc; int jp = j0 + t;
#pragma unroll
        for (int p = 0; p < 16; ++p) {
          bool ins = dp < bd[p];
          float nd = ins ? dp : bd[p];
          int   nj = ins ? jp : bj[p];
          dp = ins ? bd[p] : dp;
          jp = ins ? bj[p] : jp;
          bd[p] = nd; bj[p] = nj;
        }
        dmax = bd[15];
      }
    }
    __syncthreads();
  }
#pragma unroll
  for (int t = 0; t < 16; ++t) { listD[w][lane][t] = bd[t]; listJ[w][lane][t] = bj[t]; }
  __syncthreads();
  if (w == 0) {  // 4-way stable merge (waves cover ascending j ranges)
    int p0 = 0, p1 = 0, p2 = 0, p3 = 0;
    int* op = idx + ((size_t)b * SL + i0 + lane) * KNN;
    for (int t = 0; t < KNN; ++t) {
      float d0 = listD[0][lane][p0], d1 = listD[1][lane][p1];
      float d2 = listD[2][lane][p2], d3 = listD[3][lane][p3];
      int sel = 0; float dm = d0;
      if (d1 < dm) { dm = d1; sel = 1; }
      if (d2 < dm) { dm = d2; sel = 2; }
      if (d3 < dm) { dm = d3; sel = 3; }
      int jv;
      if      (sel == 0) jv = listJ[0][lane][p0++];
      else if (sel == 1) jv = listJ[1][lane][p1++];
      else if (sel == 2) jv = listJ[2][lane][p2++];
      else               jv = listJ[3][lane][p3++];
      op[t] = jv;
    }
  }
}

// ---------------- GEMM + bias -> bf16 (fc1)
__global__ __launch_bounds__(32) void gemm_bias(const __bf16* A, const __bf16* Wsw,
                                                const float* bias, __bf16* out) {
  int lane = threadIdx.x, ll = lane & 15, hi = lane >> 4;
  int m0 = blockIdx.x * 32;
  int n0 = blockIdx.y * 32;
  int nt0 = n0 >> 4;
  v8f c00 = {}, c01 = {}, c10 = {}, c11 = {};
#pragma unroll
  for (int kt = 0; kt < 8; ++kt) {
    v16bf a0 = load_afrag(A, m0, kt, lane);
    v16bf a1 = load_afrag(A, m0 + 16, kt, lane);
    v16bf b0 = load_bfrag(Wsw, kt, nt0, lane);
    v16bf b1 = load_bfrag(Wsw, kt, nt0 + 1, lane);
    c00 = wmma_bf16(a0, b0, c00);
    c01 = wmma_bf16(a0, b1, c01);
    c10 = wmma_bf16(a1, b0, c10);
    c11 = wmma_bf16(a1, b1, c11);
  }
  float bA = bias[n0 + ll], bB = bias[n0 + 16 + ll];
#pragma unroll
  for (int r = 0; r < 8; ++r) {
    int row0 = m0 + r + 8 * hi, row1 = row0 + 16;
    out[(size_t)row0 * DIM + n0 + ll]      = (__bf16)(c00[r] + bA);
    out[(size_t)row0 * DIM + n0 + 16 + ll] = (__bf16)(c01[r] + bB);
    out[(size_t)row1 * DIM + n0 + ll]      = (__bf16)(c10[r] + bA);
    out[(size_t)row1 * DIM + n0 + 16 + ll] = (__bf16)(c11[r] + bB);
  }
}

// ---------------- fused q/k/v GEMMs (A loaded once)
__global__ __launch_bounds__(32) void gemm_qkv(const __bf16* A, const __bf16* Wq,
                                               const __bf16* Wk, const __bf16* Wv,
                                               __bf16* Q, __bf16* Kf, __bf16* V) {
  int lane = threadIdx.x, ll = lane & 15, hi = lane >> 4;
  int m0 = blockIdx.x * 32;
  int n0 = blockIdx.y * 32;
  int nt0 = n0 >> 4;
  v8f q00 = {}, q01 = {}, q10 = {}, q11 = {};
  v8f k00 = {}, k01 = {}, k10 = {}, k11 = {};
  v8f v00 = {}, v01 = {}, v10 = {}, v11 = {};
#pragma unroll
  for (int kt = 0; kt < 8; ++kt) {
    v16bf a0 = load_afrag(A, m0, kt, lane);
    v16bf a1 = load_afrag(A, m0 + 16, kt, lane);
    v16bf b0 = load_bfrag(Wq, kt, nt0, lane);
    v16bf b1 = load_bfrag(Wq, kt, nt0 + 1, lane);
    q00 = wmma_bf16(a0, b0, q00); q01 = wmma_bf16(a0, b1, q01);
    q10 = wmma_bf16(a1, b0, q10); q11 = wmma_bf16(a1, b1, q11);
    b0 = load_bfrag(Wk, kt, nt0, lane);
    b1 = load_bfrag(Wk, kt, nt0 + 1, lane);
    k00 = wmma_bf16(a0, b0, k00); k01 = wmma_bf16(a0, b1, k01);
    k10 = wmma_bf16(a1, b0, k10); k11 = wmma_bf16(a1, b1, k11);
    b0 = load_bfrag(Wv, kt, nt0, lane);
    b1 = load_bfrag(Wv, kt, nt0 + 1, lane);
    v00 = wmma_bf16(a0, b0, v00); v01 = wmma_bf16(a0, b1, v01);
    v10 = wmma_bf16(a1, b0, v10); v11 = wmma_bf16(a1, b1, v11);
  }
#pragma unroll
  for (int r = 0; r < 8; ++r) {
    int row0 = m0 + r + 8 * hi, row1 = row0 + 16;
    size_t o0a = (size_t)row0 * DIM + n0 + ll, o0b = o0a + 16;
    size_t o1a = (size_t)row1 * DIM + n0 + ll, o1b = o1a + 16;
    Q[o0a] = (__bf16)q00[r]; Q[o0b] = (__bf16)q01[r];
    Q[o1a] = (__bf16)q10[r]; Q[o1b] = (__bf16)q11[r];
    Kf[o0a] = (__bf16)k00[r]; Kf[o0b] = (__bf16)k01[r];
    Kf[o1a] = (__bf16)k10[r]; Kf[o1b] = (__bf16)k11[r];
    V[o0a] = (__bf16)v00[r]; V[o0b] = (__bf16)v01[r];
    V[o1a] = (__bf16)v10[r]; V[o1b] = (__bf16)v11[r];
  }
}

// ---------------- fused gather + relation-MLP + softmax + weighted sum (1 wave / point)
__global__ __launch_bounds__(32) void attn_kernel(
    const __bf16* Q, const __bf16* Kf, const __bf16* V, const int* idx,
    const __bf16* G1, const __bf16* G2, const float* g1b, const float* g2b,
    __bf16* res) {
  __shared__ __bf16 Xs[16 * DIM];
  __shared__ __bf16 Vs[16 * DIM];
  int lane = threadIdx.x, ll = lane & 15, hi = lane >> 4;
  int p = blockIdx.x;                       // global point row
  int nb = idx[(size_t)p * KNN + ll];       // neighbor for A-row ll
  size_t nrow = (size_t)(p / SL) * SL + nb;

  {  // stage gathered V rows into LDS (coalesced per-row halves, 256B aligned)
    const uint4* src = (const uint4*)(V + nrow * DIM + hi * 128);
    uint4* dst = (uint4*)(Vs + ll * DIM + hi * 128);
#if HAVE_ASYNC_LDS
#pragma unroll
    for (int t = 0; t < 16; ++t) async_g2l_b128(src + t, dst + t);
#else
#pragma unroll
    for (int t = 0; t < 16; ++t) dst[t] = src[t];
#endif
  }

  // relation = q - kg, held as 8 resident A fragments
  v16bf relA[8];
  const __bf16* qrow = Q + (size_t)p * DIM;
  const __bf16* krow = Kf + nrow * DIM;
#pragma unroll
  for (int kt = 0; kt < 8; ++kt) {
    int c0 = kt * 32 + 8 * hi;
    Frag16 fq, fk, fr;
    fq.u[0] = *(const uint4*)(qrow + c0); fq.u[1] = *(const uint4*)(qrow + c0 + 16);
    fk.u[0] = *(const uint4*)(krow + c0); fk.u[1] = *(const uint4*)(krow + c0 + 16);
#pragma unroll
    for (int e = 0; e < 16; ++e)
      fr.v[e] = (__bf16)((float)fq.v[e] - (float)fk.v[e]);
    relA[kt] = fr.v;
  }

  // GEMM1: X = relu(rel @ g1 + b1) -> LDS (row-major bf16)
  for (int nt = 0; nt < 16; ++nt) {
    float bb = g1b[nt * 16 + ll];
    v8f acc;
#pragma unroll
    for (int e = 0; e < 8; ++e) acc[e] = bb;
#pragma unroll
    for (int kt = 0; kt < 8; ++kt)
      acc = wmma_bf16(relA[kt], load_bfrag(G1, kt, nt, lane), acc);
#pragma unroll
    for (int r = 0; r < 8; ++r)
      Xs[(r + 8 * hi) * DIM + nt * 16 + ll] = (__bf16)fmaxf(acc[r], 0.0f);
  }
#if HAVE_ASYNC_LDS
  async_wait0();  // V staging overlapped with GEMM1
#endif
  __syncthreads();

  // GEMM2 + softmax over K (column-wise, 16 rows) + weighted sum with V
  for (int nt = 0; nt < 16; ++nt) {
    float bb = g2b[nt * 16 + ll];
    v8f acc;
#pragma unroll
    for (int e = 0; e < 8; ++e) acc[e] = bb;
#pragma unroll
    for (int kt = 0; kt < 8; ++kt) {
      Frag16 fx;
      const __bf16* xr = Xs + ll * DIM + kt * 32 + 8 * hi;
      fx.u[0] = *(const uint4*)xr;
      fx.u[1] = *(const uint4*)(xr + 16);
      acc = wmma_bf16(fx.v, load_bfrag(G2, kt, nt, lane), acc);
    }
    float t[8];
    float m = -3.4e38f;
#pragma unroll
    for (int r = 0; r < 8; ++r) { t[r] = acc[r] * 0.0625f; m = fmaxf(m, t[r]); }
    m = fmaxf(m, __shfl_xor(m, 16, 32));
    float s = 0.f;
#pragma unroll
    for (int r = 0; r < 8; ++r) { t[r] = __expf(t[r] - m); s += t[r]; }
    s += __shfl_xor(s, 16, 32);
    float inv = 1.0f / s;
    float part = 0.f;
#pragma unroll
    for (int r = 0; r < 8; ++r) {
      float v = (float)Vs[(r + 8 * hi) * DIM + nt * 16 + ll];
      part += t[r] * inv * v;
    }
    part += __shfl_xor(part, 16, 32);
    if (hi == 0) res[(size_t)p * DIM + nt * 16 + ll] = (__bf16)part;
  }
}

// ---------------- fc2 + bias + residual feat -> f32 out
__global__ __launch_bounds__(32) void gemm_fc2(const __bf16* A, const __bf16* Wsw,
                                               const float* bias, const float* x,
                                               float* out) {
  int lane = threadIdx.x, ll = lane & 15, hi = lane >> 4;
  int m0 = blockIdx.x * 32;
  int n0 = blockIdx.y * 32;
  int nt0 = n0 >> 4;
  v8f c00 = {}, c01 = {}, c10 = {}, c11 = {};
#pragma unroll
  for (int kt = 0; kt < 8; ++kt) {
    v16bf a0 = load_afrag(A, m0, kt, lane);
    v16bf a1 = load_afrag(A, m0 + 16, kt, lane);
    v16bf b0 = load_bfrag(Wsw, kt, nt0, lane);
    v16bf b1 = load_bfrag(Wsw, kt, nt0 + 1, lane);
    c00 = wmma_bf16(a0, b0, c00);
    c01 = wmma_bf16(a0, b1, c01);
    c10 = wmma_bf16(a1, b0, c10);
    c11 = wmma_bf16(a1, b1, c11);
  }
  float bA = bias[n0 + ll], bB = bias[n0 + 16 + ll];
#pragma unroll
  for (int r = 0; r < 8; ++r) {
    int row0 = m0 + r + 8 * hi, row1 = row0 + 16;
    int nA = n0 + ll, nB = n0 + 16 + ll;
    out[(size_t)row0 * DIM + nA] = c00[r] + bA + x[(size_t)row0 * XST + 3 + nA];
    out[(size_t)row0 * DIM + nB] = c01[r] + bB + x[(size_t)row0 * XST + 3 + nB];
    out[(size_t)row1 * DIM + nA] = c10[r] + bA + x[(size_t)row1 * XST + 3 + nA];
    out[(size_t)row1 * DIM + nB] = c11[r] + bB + x[(size_t)row1 * XST + 3 + nB];
  }
}

extern "C" void kernel_launch(void* const* d_in, const int* in_sizes, int n_in,
                              void* d_out, int out_size, void* d_ws, size_t ws_size,
                              hipStream_t stream) {
  const float* x     = (const float*)d_in[0];
  const float* fc1_w = (const float*)d_in[1];
  const float* fc1_b = (const float*)d_in[2];
  const float* fc2_w = (const float*)d_in[3];
  const float* fc2_b = (const float*)d_in[4];
  const float* wq_w  = (const float*)d_in[5];
  const float* wk_w  = (const float*)d_in[6];
  const float* wv_w  = (const float*)d_in[7];
  const float* g1_w  = (const float*)d_in[8];
  const float* g1_b  = (const float*)d_in[9];
  const float* g2_w  = (const float*)d_in[10];
  const float* g2_b  = (const float*)d_in[11];

  char* ws = (char*)d_ws;
  size_t off = 0;
  auto carve = [&](size_t bytes) -> char* {
    char* p = ws + off;
    off += (bytes + 255) & ~(size_t)255;
    return p;
  };
  __bf16* wsw   = (__bf16*)carve((size_t)7 * DIMSQ * 2);
  float*  sq    = (float*)carve((size_t)NROW * 4);
  int*    idx   = (int*)carve((size_t)NROW * KNN * 4);
  __bf16* featB = (__bf16*)carve((size_t)NROW * DIM * 2);
  __bf16* hB    = (__bf16*)carve((size_t)NROW * DIM * 2);
  __bf16* qB    = (__bf16*)carve((size_t)NROW * DIM * 2);
  __bf16* kB    = (__bf16*)carve((size_t)NROW * DIM * 2);
  __bf16* vB    = (__bf16*)carve((size_t)NROW * DIM * 2);
  __bf16* resB  = (__bf16*)carve((size_t)NROW * DIM * 2);

  float* posOut = (float*)d_out;                       // (B,S,3)
  float* outF   = (float*)d_out + (size_t)NROW * 3;    // (B,S,D)

  // weight order in wsw: fc1, wq, wk, wv, g1, g2, fc2
  prep_weights<<<dim3(128, 7), 32, 0, stream>>>(fc1_w, wq_w, wk_w, wv_w, g1_w, g2_w,
                                                fc2_w, wsw);
  prep_feat<<<NROW, 32, 0, stream>>>(x, sq, featB, posOut);
  knn_kernel<<<NROW / 32, 128, 0, stream>>>(x, sq, idx);
  gemm_bias<<<dim3(NROW / 32, 8), 32, 0, stream>>>(featB, wsw + 0 * DIMSQ, fc1_b, hB);
  gemm_qkv<<<dim3(NROW / 32, 8), 32, 0, stream>>>(hB, wsw + 1 * DIMSQ, wsw + 2 * DIMSQ,
                                                  wsw + 3 * DIMSQ, qB, kB, vB);
  attn_kernel<<<NROW, 32, 0, stream>>>(qB, kB, vB, idx, wsw + 4 * DIMSQ,
                                       wsw + 5 * DIMSQ, g1_b, g2_b, resB);
  gemm_fc2<<<dim3(NROW / 32, 8), 32, 0, stream>>>(resB, wsw + 6 * DIMSQ, fc2_b, x,
                                                  outF);
}
